// involution_81106162418291
// MI455X (gfx1250) — matrix operand order repaired and checked by reading, not compiled, not run on hardware
//
#include <hip/hip_runtime.h>

typedef __attribute__((ext_vector_type(16))) _Float16 v16h;
typedef __attribute__((ext_vector_type(8)))  _Float16 v8h;
typedef __attribute__((ext_vector_type(8)))  float    v8f;
typedef __attribute__((ext_vector_type(4)))  float    v4f;

#define Bsz 4
#define Cc  256
#define Hh  56
#define Ww  56
#define HW  3136
#define CR  64
#define Gg  64
#define KK  7
#define K2  49
#define HALO 22                   // 16 + 6 halo columns
#define XSN (4 * KK * HALO)       // 616 floats staged per wave

// ---- kernel 0: w2 (3136x64 f32) -> padded f16 (64g x 64row x 64cr); b2 -> padded f32 ----------
__global__ __launch_bounds__(256) void k_cvt_w2(const float* __restrict__ w2,
                                                const float* __restrict__ b2,
                                                _Float16* __restrict__ w2h,
                                                float* __restrict__ b2p) {
  const int gid = blockIdx.x * 256 + threadIdx.x;       // 64*64*64 = 262144
  const int cr = gid & 63;
  const int m  = (gid >> 6) & 63;
  const int g  = gid >> 12;
  const float v = (m < K2) ? w2[(size_t)(g * K2 + m) * CR + cr] : 0.f;
  w2h[gid] = (_Float16)v;                               // row (g*64+m), col cr
  if (gid < Gg * 64) {                                  // first 4096 threads pad the bias
    const int gb = gid >> 6, row = gid & 63;
    b2p[gid] = (row < K2) ? b2[gb * K2 + row] : 0.f;
  }
}

// ---- kernel 1: y = relu(BN(w1 @ x + b1)) -> f16, layout [b][pixel][cr] ------------------------
__global__ __launch_bounds__(256) void k_conv1(
    const float* __restrict__ x, const float* __restrict__ w1,
    const float* __restrict__ b1, const float* __restrict__ gamma,
    const float* __restrict__ beta, const float* __restrict__ bmean,
    const float* __restrict__ bvar, _Float16* __restrict__ yh)
{
  __shared__ float xs[64][64];     // [c_local][pixel]
  __shared__ float ws[64][65];     // [cr][c_local], padded vs bank conflicts
  const int tid = threadIdx.x;
  const int b   = blockIdx.x / (HW / 64);
  const int p0  = (blockIdx.x % (HW / 64)) * 64;
  const int cr  = tid & 63;
  const int pq  = tid >> 6;        // 0..3, pixel phase

  float acc[16];
  #pragma unroll
  for (int i = 0; i < 16; ++i) acc[i] = 0.f;

  for (int cc = 0; cc < Cc / 64; ++cc) {
    __syncthreads();
    #pragma unroll
    for (int j = 0; j < 16; ++j) {
      const int idx = tid + j * 256;
      const int a = idx >> 6, q = idx & 63;
      xs[a][q] = x[(size_t)(b * Cc + cc * 64 + a) * HW + p0 + q];
      ws[a][q] = w1[(size_t)a * Cc + cc * 64 + q];
    }
    __syncthreads();
    for (int c = 0; c < 64; ++c) {
      const float wv = ws[cr][c];
      #pragma unroll
      for (int i = 0; i < 16; ++i)
        acc[i] = fmaf(wv, xs[c][pq + 4 * i], acc[i]);
    }
  }
  const float b1c = b1[cr];
  const float inv = gamma[cr] * rsqrtf(bvar[cr] + 1e-5f);
  const float sh  = beta[cr] - bmean[cr] * inv;
  #pragma unroll
  for (int i = 0; i < 16; ++i) {
    float v = (acc[i] + b1c) * inv + sh;
    v = fmaxf(v, 0.f);
    yh[(size_t)(b * HW + p0 + pq + 4 * i) * CR + cr] = (_Float16)v;
  }
}

// ---- kernel 2: per-wave WMMA weight generation + LDS-staged involution MAC --------------------
// wave task = (b, g, row h, 16-wide col tile w0 in {0,14,28,40}; tiles overlap, dup writes are
// bitwise identical). weight tile (49x16) = w2h_g @ y(64x16) + b2pad via v_wmma_f32_16x16x32_f16.
__global__ __launch_bounds__(256) void k_invo(
    const float* __restrict__ x, const _Float16* __restrict__ yh,
    const _Float16* __restrict__ w2h, const float* __restrict__ b2p,
    float* __restrict__ out)
{
  __shared__ float wgt[8][64][16];          // per-wave weight tile
  __shared__ float xs[8][4][KK][HALO];      // per-wave zero-padded x halo (4 ch x 7 rows x 22 col)
  const int lane  = threadIdx.x & 31;
  const int wslot = threadIdx.x >> 5;
  const int wid   = blockIdx.x * 8 + wslot; // 0 .. 57343
  const int tw    = wid & 3;
  const int h     = (wid >> 2) % Hh;
  const int g     = (wid / (4 * Hh)) & 63;
  const int b     = wid / (4 * Hh * Gg);
  const int w0    = tw * 14 - ((tw == 3) ? 2 : 0);   // 0,14,28,40
  const int p0    = h * Ww + w0;
  const int hilo  = lane >> 4;              // lane half
  const int lm    = lane & 15;

  // ---- cooperative halo stage: x[g*4 .. g*4+3][h-3..h+3][w0-3..w0+18] -> LDS (zeros outside) --
  {
    const float* xg = x + (size_t)(b * Cc + g * 4) * HW;
    #pragma unroll
    for (int j = 0; j < (XSN + 31) / 32; ++j) {
      const int idx = lane + 32 * j;
      if (idx < XSN) {
        const int c   = idx / (KK * HALO);
        const int rem = idx % (KK * HALO);
        const int dy  = rem / HALO;
        const int dx  = rem % HALO;
        const int iy  = h + dy - 3;
        const int ix  = w0 + dx - 3;
        float v = 0.f;
        if (((unsigned)iy < (unsigned)Hh) && ((unsigned)ix < (unsigned)Ww))
          v = xg[(size_t)c * HW + iy * Ww + ix];
        xs[wslot][c][dy][dx] = v;
      }
    }
  }

  // ---- B fragments (activations y): lane holds pixel p0+lm, K = two 8-cr runs at hilo*8, hilo*8+16 ------
  v16h bf[2];
  {
    const _Float16* yp = yh + (size_t)(b * HW + p0 + lm) * CR + hilo * 8;
    #pragma unroll
    for (int s = 0; s < 2; ++s) {
      v8h lo = *(const v8h*)(yp + s * 32);
      v8h hi = *(const v8h*)(yp + s * 32 + 16);
      #pragma unroll
      for (int i = 0; i < 8; ++i) { bf[s][i] = lo[i]; bf[s][8 + i] = hi[i]; }
    }
  }

  v8f acc[4];
  #pragma unroll
  for (int t = 0; t < 4; ++t) {
    // C init = padded bias broadcast (rows t*16 + hilo*8 + 0..7, contiguous, unconditional)
    {
      const float* bp = b2p + g * 64 + t * 16 + hilo * 8;
      v4f blo = *(const v4f*)(bp);
      v4f bhi = *(const v4f*)(bp + 4);
      #pragma unroll
      for (int r = 0; r < 4; ++r) { acc[t][r] = blo[r]; acc[t][4 + r] = bhi[r]; }
    }
    #pragma unroll
    for (int s = 0; s < 2; ++s) {
      // A fragment: row M = t*16 + lm; two 8-K runs at (hilo*8) and (hilo*8 + 16)
      const _Float16* ap = w2h + (size_t)(g * 64 + t * 16 + lm) * CR + s * 32 + hilo * 8;
      v8h lo = *(const v8h*)(ap);
      v8h hi = *(const v8h*)(ap + 16);
      v16h af;
      #pragma unroll
      for (int i = 0; i < 8; ++i) { af[i] = lo[i]; af[8 + i] = hi[i]; }
      acc[t] = __builtin_amdgcn_wmma_f32_16x16x32_f16(false, af, false, bf[s],
                                                      (short)0, acc[t], false, false);
    }
    // D layout -> LDS: VGPR r holds row t*16 + hilo*8 + r for pixel lm
    #pragma unroll
    for (int r = 0; r < 8; ++r)
      wgt[wslot][t * 16 + hilo * 8 + r][lm] = acc[t][r];
  }

  // ---- involution MAC, fully branchless from LDS: 2 channels x 1 pixel per lane --------------
  const int cA = hilo * 2;                  // channels cA, cA+1 within the group
  float a0 = 0.f, a1 = 0.f;
  #pragma unroll
  for (int ky = 0; ky < KK; ++ky) {
    #pragma unroll
    for (int kx = 0; kx < KK; ++kx) {
      const float wv = wgt[wslot][ky * KK + kx][lm];
      const float x0 = xs[wslot][cA][ky][lm + kx];
      const float x1 = xs[wslot][cA + 1][ky][lm + kx];
      a0 = fmaf(wv, x0, a0);
      a1 = fmaf(wv, x1, a1);
    }
  }
  const int c0 = g * 4 + cA;
  out[(size_t)(b * Cc + c0) * HW + p0 + lm]     = a0;
  out[(size_t)(b * Cc + c0 + 1) * HW + p0 + lm] = a1;
}

extern "C" void kernel_launch(void* const* d_in, const int* in_sizes, int n_in,
                              void* d_out, int out_size, void* d_ws, size_t ws_size,
                              hipStream_t stream) {
  const float* x     = (const float*)d_in[0];
  const float* w1    = (const float*)d_in[1];
  const float* b1    = (const float*)d_in[2];
  const float* gamma = (const float*)d_in[3];
  const float* beta  = (const float*)d_in[4];
  const float* bmean = (const float*)d_in[5];
  const float* bvar  = (const float*)d_in[6];
  const float* w2    = (const float*)d_in[7];
  const float* b2    = (const float*)d_in[8];
  float* out = (float*)d_out;

  char* ws = (char*)d_ws;
  _Float16* yh  = (_Float16*)ws;                                  // 4*3136*64 f16 = 1,605,632 B
  _Float16* w2h = (_Float16*)(ws + (size_t)Bsz * HW * CR * 2);    // 64*64*64 f16 =   524,288 B
  float*    b2p = (float*)(ws + (size_t)Bsz * HW * CR * 2 + (size_t)Gg * 64 * CR * 2); // 16 KB

  k_cvt_w2<<<(Gg * 64 * CR) / 256, 256, 0, stream>>>(w2, b2, w2h, b2p);
  k_conv1<<<Bsz * (HW / 64), 256, 0, stream>>>(x, w1, b1, gamma, beta, bmean, bvar, yh);
  k_invo<<<(Bsz * Gg * Hh * 4) / 8, 256, 0, stream>>>(x, yh, w2h, b2p, out);
}
